// MistralQuantizedMLP_43035572306187
// MI455X (gfx1250) — matrix operand order repaired and checked by reading, not compiled, not run on hardware
//
#include <hip/hip_runtime.h>
#include <cstdint>
#include <cstddef>

typedef __attribute__((ext_vector_type(16))) _Float16 v16h;
typedef __attribute__((ext_vector_type(8)))  _Float16 v8h;
typedef __attribute__((ext_vector_type(8)))  float    v8f;
typedef __attribute__((ext_vector_type(4)))  int      v4i;

#define M_TOK 4096   // B*S
#define HDIM  4096
#define IDIM  14336

// ---------------- CDNA5 async copy helpers ----------------
// GLOBAL_LOAD_ASYNC_TO_LDS_B128 (ISA 15.18.3 op 98): VDST = per-lane LDS byte
// address, VADDR = 64-bit global address, tracked with ASYNCcnt.
__device__ __forceinline__ void async_ld_b128(uint32_t lds_addr, const void* gaddr) {
    asm volatile("global_load_async_to_lds_b128 %0, %1, off"
                 :: "v"(lds_addr), "v"((uint64_t)(uintptr_t)gaddr)
                 : "memory");
}
__device__ __forceinline__ void wait_async0() {
#if __has_builtin(__builtin_amdgcn_s_wait_asynccnt)
    __builtin_amdgcn_s_wait_asynccnt(0);
#else
    asm volatile("s_wait_asynccnt 0" ::: "memory");
#endif
}
__device__ __forceinline__ uint32_t lds_off(const void* p) {
    return (uint32_t)(uintptr_t)p;   // low 32 bits of a flat LDS pointer = LDS offset
}

// ---------------- kernel 1: x fp32 -> fp16 ----------------
__global__ __launch_bounds__(256) void k_cvt_f32_f16(const float* __restrict__ src,
                                                     _Float16* __restrict__ dst) {
    size_t i = ((size_t)blockIdx.x * 256 + threadIdx.x) * 8;
    float4 a = *(const float4*)(src + i);
    float4 b = *(const float4*)(src + i + 4);
    v8h o;
    o[0] = (_Float16)a.x; o[1] = (_Float16)a.y; o[2] = (_Float16)a.z; o[3] = (_Float16)a.w;
    o[4] = (_Float16)b.x; o[5] = (_Float16)b.y; o[6] = (_Float16)b.z; o[7] = (_Float16)b.w;
    *(v8h*)(dst + i) = o;
}

// --------------- WMMA fragment loaders (wave32 layouts, ISA 7.12.2) ---------------
__device__ __forceinline__ v16h load_a_frag(const _Float16* __restrict__ s, int lane) {
    const int m  = lane & 15;
    const int hi = lane >> 4;
    const _Float16* r = s + m * 32;
    v8h lo = *(const v8h*)(r + 8 * hi);
    v8h hh = *(const v8h*)(r + 16 + 8 * hi);
    v16h a;
#pragma unroll
    for (int j = 0; j < 8; ++j) { a[j] = lo[j]; a[j + 8] = hh[j]; }
    return a;
}
__device__ __forceinline__ v16h load_b_frag(const _Float16* __restrict__ s, int lane) {
    const int n  = lane & 15;
    const int kh = lane >> 4;
    const _Float16* r = s + n * 32;
    v8h lo = *(const v8h*)(r + 8 * kh);
    v8h hh = *(const v8h*)(r + 16 + 8 * kh);
    v16h b;
#pragma unroll
    for (int j = 0; j < 8; ++j) { b[j] = lo[j]; b[j + 8] = hh[j]; }
    return b;
}

#define WMMA_F16(A, B, C) \
    __builtin_amdgcn_wmma_f32_16x16x32_f16(false, (A), false, (B), (short)0, (C), false, false)

__device__ __forceinline__ v8h cvt8(v4i a, v4i b) {
    v8h o;
#pragma unroll
    for (int j = 0; j < 4; ++j) {
        o[j]     = (_Float16)(float)a[j];
        o[j + 4] = (_Float16)(float)b[j];
    }
    return o;
}

// --------------- kernel 2: fused gate/up GEMM + SwiGLU -> h (f16) ---------------
// grid = (IDIM/64, M_TOK/128), 256 threads, double-buffered LDS, async x staging
__global__ __launch_bounds__(256) void k_gateup(
    const _Float16* __restrict__ x16,
    const int* __restrict__ gw, const float* __restrict__ gsc,
    const int* __restrict__ uw, const float* __restrict__ usc,
    _Float16* __restrict__ h16)
{
    __shared__ __attribute__((aligned(16))) _Float16 sX[2][128 * 32];
    __shared__ __attribute__((aligned(16))) _Float16 sG[2][64 * 32];
    __shared__ __attribute__((aligned(16))) _Float16 sU[2][64 * 32];

    const int tid  = threadIdx.x;
    const int lane = tid & 31;
    const int wid  = tid >> 5;
    const int wm   = (wid & 3) * 32;
    const int wn   = (wid >> 2) * 32;
    const int nBase = blockIdx.x * 64;
    const int mBase = blockIdx.y * 128;

    v8f accG[2][2], accU[2][2];
#pragma unroll
    for (int i = 0; i < 2; ++i)
#pragma unroll
        for (int j = 0; j < 2; ++j) { accG[i][j] = {}; accU[i][j] = {}; }

    // staging coordinates (2 x-tile segments + 1 weight segment per thread)
    const int xr0 = tid >> 2,         xc0 = (tid & 3) * 8;
    const int xr1 = (tid + 256) >> 2, xc1 = (tid & 3) * 8;
    const int wrow = tid >> 2,        wc8 = (tid & 3) * 8;

    v4i pg0, pg1, pu0, pu1;   // weight prefetch registers

    auto issue_x = [&](int buf, int k0) {
        async_ld_b128(lds_off(&sX[buf][xr0 * 32 + xc0]),
                      x16 + (size_t)(mBase + xr0) * HDIM + k0 + xc0);
        async_ld_b128(lds_off(&sX[buf][xr1 * 32 + xc1]),
                      x16 + (size_t)(mBase + xr1) * HDIM + k0 + xc1);
    };
    auto issue_w = [&](int k0) {
        const v4i* g = (const v4i*)(gw + (size_t)(nBase + wrow) * HDIM + k0 + wc8);
        const v4i* u = (const v4i*)(uw + (size_t)(nBase + wrow) * HDIM + k0 + wc8);
        pg0 = __builtin_nontemporal_load(g);       // NT: read-once weight stream,
        pg1 = __builtin_nontemporal_load(g + 1);   // keep x16 resident in L2
        pu0 = __builtin_nontemporal_load(u);
        pu1 = __builtin_nontemporal_load(u + 1);
    };
    auto store_w = [&](int buf) {
        *(v8h*)&sG[buf][wrow * 32 + wc8] = cvt8(pg0, pg1);
        *(v8h*)&sU[buf][wrow * 32 + wc8] = cvt8(pu0, pu1);
    };

    // prologue: stage tile 0
    issue_x(0, 0);
    issue_w(0);
    store_w(0);
    wait_async0();
    __syncthreads();

    const int NIT = HDIM / 32;
    for (int it = 0; it < NIT; ++it) {
        const int cur = it & 1;
        const int nxt = cur ^ 1;
        if (it + 1 < NIT) {             // issue next tile (overlaps with WMMAs)
            issue_x(nxt, (it + 1) * 32);
            issue_w((it + 1) * 32);
        }

        const v16h a0  = load_a_frag(&sX[cur][(wm +  0) * 32], lane);
        const v16h a1  = load_a_frag(&sX[cur][(wm + 16) * 32], lane);
        const v16h bg0 = load_b_frag(&sG[cur][(wn +  0) * 32], lane);
        const v16h bg1 = load_b_frag(&sG[cur][(wn + 16) * 32], lane);
        const v16h bu0 = load_b_frag(&sU[cur][(wn +  0) * 32], lane);
        const v16h bu1 = load_b_frag(&sU[cur][(wn + 16) * 32], lane);

        accG[0][0] = WMMA_F16(a0, bg0, accG[0][0]);
        accG[0][1] = WMMA_F16(a0, bg1, accG[0][1]);
        accG[1][0] = WMMA_F16(a1, bg0, accG[1][0]);
        accG[1][1] = WMMA_F16(a1, bg1, accG[1][1]);
        accU[0][0] = WMMA_F16(a0, bu0, accU[0][0]);
        accU[0][1] = WMMA_F16(a0, bu1, accU[0][1]);
        accU[1][0] = WMMA_F16(a1, bu0, accU[1][0]);
        accU[1][1] = WMMA_F16(a1, bu1, accU[1][1]);

        if (it + 1 < NIT) {
            store_w(nxt);
            wait_async0();
        }
        __syncthreads();
    }

    // epilogue: per-channel scale, SiLU(gate)*up, store f16 h
    const int hi = lane >> 4;
#pragma unroll
    for (int nt = 0; nt < 2; ++nt) {
        const int n = nBase + wn + nt * 16 + (lane & 15);
        const float gscale = gsc[n];
        const float uscale = usc[n];
#pragma unroll
        for (int mt = 0; mt < 2; ++mt) {
            const int mrow = mBase + wm + mt * 16 + 8 * hi;
#pragma unroll
            for (int v = 0; v < 8; ++v) {
                const float g = accG[mt][nt][v] * gscale;
                const float u = accU[mt][nt][v] * uscale;
                const float hval = (g / (1.0f + __expf(-g))) * u;
                h16[(size_t)(mrow + v) * IDIM + n] = (_Float16)hval;
            }
        }
    }
}

// --------------- kernel 3: down GEMM: out = h @ Wd^T * scale (fp32) ---------------
// grid = (HDIM/64, M_TOK/128)
__global__ __launch_bounds__(256) void k_down(
    const _Float16* __restrict__ h16,
    const int* __restrict__ dw, const float* __restrict__ dsc,
    float* __restrict__ out)
{
    __shared__ __attribute__((aligned(16))) _Float16 sA[2][128 * 32];
    __shared__ __attribute__((aligned(16))) _Float16 sW[2][64 * 32];

    const int tid  = threadIdx.x;
    const int lane = tid & 31;
    const int wid  = tid >> 5;
    const int wm   = (wid & 3) * 32;
    const int wn   = (wid >> 2) * 32;
    const int nBase = blockIdx.x * 64;
    const int mBase = blockIdx.y * 128;

    v8f acc[2][2];
#pragma unroll
    for (int i = 0; i < 2; ++i)
#pragma unroll
        for (int j = 0; j < 2; ++j) acc[i][j] = {};

    const int xr0 = tid >> 2,         xc0 = (tid & 3) * 8;
    const int xr1 = (tid + 256) >> 2, xc1 = (tid & 3) * 8;
    const int wrow = tid >> 2,        wc8 = (tid & 3) * 8;

    v4i pw0, pw1;

    auto issue_a = [&](int buf, int k0) {
        async_ld_b128(lds_off(&sA[buf][xr0 * 32 + xc0]),
                      h16 + (size_t)(mBase + xr0) * IDIM + k0 + xc0);
        async_ld_b128(lds_off(&sA[buf][xr1 * 32 + xc1]),
                      h16 + (size_t)(mBase + xr1) * IDIM + k0 + xc1);
    };
    auto issue_w = [&](int k0) {
        const v4i* w = (const v4i*)(dw + (size_t)(nBase + wrow) * IDIM + k0 + wc8);
        pw0 = __builtin_nontemporal_load(w);
        pw1 = __builtin_nontemporal_load(w + 1);
    };
    auto store_w = [&](int buf) {
        *(v8h*)&sW[buf][wrow * 32 + wc8] = cvt8(pw0, pw1);
    };

    issue_a(0, 0);
    issue_w(0);
    store_w(0);
    wait_async0();
    __syncthreads();

    const int NIT = IDIM / 32;
    for (int it = 0; it < NIT; ++it) {
        const int cur = it & 1;
        const int nxt = cur ^ 1;
        if (it + 1 < NIT) {
            issue_a(nxt, (it + 1) * 32);
            issue_w((it + 1) * 32);
        }

        const v16h a0 = load_a_frag(&sA[cur][(wm +  0) * 32], lane);
        const v16h a1 = load_a_frag(&sA[cur][(wm + 16) * 32], lane);
        const v16h b0 = load_b_frag(&sW[cur][(wn +  0) * 32], lane);
        const v16h b1 = load_b_frag(&sW[cur][(wn + 16) * 32], lane);

        acc[0][0] = WMMA_F16(a0, b0, acc[0][0]);
        acc[0][1] = WMMA_F16(a0, b1, acc[0][1]);
        acc[1][0] = WMMA_F16(a1, b0, acc[1][0]);
        acc[1][1] = WMMA_F16(a1, b1, acc[1][1]);

        if (it + 1 < NIT) {
            store_w(nxt);
            wait_async0();
        }
        __syncthreads();
    }

    const int hi = lane >> 4;
#pragma unroll
    for (int nt = 0; nt < 2; ++nt) {
        const int n = nBase + wn + nt * 16 + (lane & 15);
        const float dscale = dsc[n];
#pragma unroll
        for (int mt = 0; mt < 2; ++mt) {
            const int mrow = mBase + wm + mt * 16 + 8 * hi;
#pragma unroll
            for (int v = 0; v < 8; ++v)
                out[(size_t)(mrow + v) * HDIM + n] = acc[mt][nt][v] * dscale;
        }
    }
}

extern "C" void kernel_launch(void* const* d_in, const int* in_sizes, int n_in,
                              void* d_out, int out_size, void* d_ws, size_t ws_size,
                              hipStream_t stream) {
    (void)in_sizes; (void)n_in; (void)out_size; (void)ws_size;

    const float* x   = (const float*)d_in[0];
    const int*   gw  = (const int*)d_in[1];
    const float* gsc = (const float*)d_in[2];
    const int*   uw  = (const int*)d_in[3];
    const float* usc = (const float*)d_in[4];
    const int*   dw  = (const int*)d_in[5];
    const float* dsc = (const float*)d_in[6];
    float*       out = (float*)d_out;

    _Float16* x16 = (_Float16*)d_ws;                       // 33.5 MB
    _Float16* h16 = x16 + (size_t)M_TOK * HDIM;            // 117 MB

    k_cvt_f32_f16<<<(M_TOK * HDIM) / (256 * 8), 256, 0, stream>>>(x, x16);
    k_gateup<<<dim3(IDIM / 64, M_TOK / 128), 256, 0, stream>>>(x16, gw, gsc, uw, usc, h16);
    k_down<<<dim3(HDIM / 64, M_TOK / 128), 256, 0, stream>>>(h16, dw, dsc, out);
}
